// EncoderLSTM_43508018709151
// MI455X (gfx1250) — matrix operand order repaired and checked by reading, not compiled, not run on hardware
//
#include <hip/hip_runtime.h>

// ---------------------------------------------------------------------------
// EncoderLSTM for MI455X (gfx1250, wave32, WMMA + TDM).
//   Phase 1: x_proj = embs @ w_ih^T + (b_ih+b_hh)  (f16 WMMA, f32 acc),
//            stored in WMMA C-fragment order.
//   Phase 2: persistent 512-step recurrence; w_hh f16 slice DMA'd into LDS
//            once per WG via tensor_load_to_lds; c-state in VGPRs;
//            grid barrier per timestep.
// ---------------------------------------------------------------------------

typedef __attribute__((ext_vector_type(16))) _Float16 v16h;
typedef __attribute__((ext_vector_type(8)))  _Float16 v8h;
typedef __attribute__((ext_vector_type(8)))  float    v8f;
typedef __attribute__((ext_vector_type(4)))  unsigned v4u;
typedef __attribute__((ext_vector_type(8)))  unsigned v8u;

#define LSTM_T    512
#define LSTM_B    32
#define LSTM_I    1024
#define LSTM_C    1024
#define LSTM_G4   4096
#define LSTM_M    (LSTM_T * LSTM_B)      // 16384

// d_ws layout (bytes)
#define XP_BYTES   ((size_t)LSTM_M * LSTM_G4 * 4)        // 256 MB  f32 x_proj (fragment order)
#define EMB_BYTES  ((size_t)LSTM_M * LSTM_I * 2)         //  32 MB  f16 embs
#define WIH_BYTES  ((size_t)LSTM_G4 * LSTM_I * 2)        //   8 MB  f16 w_ih
#define WHH_BYTES  ((size_t)LSTM_G4 * LSTM_I * 2)        //   8 MB  f16 w_hh (fragment order)
#define HBUF_BYTES ((size_t)2 * LSTM_B * LSTM_C * 2)     // 128 KB  f16 h ping-pong

// Per-WG w_hh slice in LDS: [gate][kstep][lane][16 halves] = 128 KB (dynamic LDS)
#define BSH_HALVES (4 * 32 * 32 * 16)
#define LSTM_SMEM  (BSH_HALVES * 2)

// --- WMMA fragment loaders (contiguous k = 16*(lane/16)+s in both A and B) --------
// A 16x32 f16: lane<16 -> row = base+lane,    K = k0..k0+15
//              lane>=16-> row = base+lane-16, K = k0+16..k0+31
__device__ __forceinline__ v16h load_afrag(const _Float16* __restrict__ base,
                                           int row0, int k0, int lane, int ld) {
  const int r = row0 + (lane & 15);
  const _Float16* p = base + (size_t)r * ld + k0 + ((lane >= 16) ? 16 : 0);
  union { v16h v; v8h h[2]; } u;
  u.h[0] = *(const v8h*)(p);
  u.h[1] = *(const v8h*)(p + 8);
  return u.v;
}
// B 32x16 f16 (column n stored row-major over K in memory):
// lane<16 -> n = n0+lane, K = k0..k0+15 ; lane>=16 -> n = n0+lane-16, K = k0+16..k0+31
__device__ __forceinline__ v16h load_bfrag(const _Float16* __restrict__ base,
                                           int n0, int k0, int lane, int ld) {
  const int n = n0 + (lane & 15);
  const _Float16* p = base + (size_t)n * ld + k0 + ((lane >= 16) ? 16 : 0);
  union { v16h v; v8h h[2]; } u;
  u.h[0] = *(const v8h*)(p);
  u.h[1] = *(const v8h*)(p + 8);
  return u.v;
}

__device__ __forceinline__ v8f wmma_f16(v16h a, v16h b, v8f c) {
  return __builtin_amdgcn_wmma_f32_16x16x32_f16(false, a, false, b, (short)0, c,
                                                false, false);
}

// --- Phase 0a: generic fp32 -> f16 -------------------------------------------
__global__ void lstm_cvt_f16(const float* __restrict__ s, _Float16* __restrict__ d, int n) {
  int i = blockIdx.x * blockDim.x + threadIdx.x;
  int stride = gridDim.x * blockDim.x;
  for (; i < n; i += stride) d[i] = (_Float16)s[i];
}

// --- Phase 0b: w_hh fp32 -> f16 in B-fragment order --------------------------
// dst[(((nt*4+g)*32+k)*32+ln)*16+hh] so each WG's slice is one contiguous 128KB
// block, ready for a single tensor_load_to_lds.
__global__ void lstm_cvt_whh_frag(const float* __restrict__ src, _Float16* __restrict__ dst) {
  const int idx = blockIdx.x * blockDim.x + threadIdx.x;   // 64*65536 total
  const int hh = idx & 15;
  const int ln = (idx >> 4) & 31;
  const int k  = (idx >> 9) & 31;
  const int g  = (idx >> 14) & 3;
  const int nt = idx >> 16;
  const int nl = ln & 15;
  const int kk = (ln < 16) ? hh : (16 + hh);
  dst[idx] = (_Float16)src[(size_t)(g * LSTM_C + nt * 16 + nl) * LSTM_C + (k * 32 + kk)];
}

// h0 broadcast into ping-pong buffer 0; zero the grid-barrier counter.
__global__ void lstm_init(const float* __restrict__ h0, _Float16* __restrict__ hbuf,
                          unsigned* __restrict__ cnt) {
  int i = blockIdx.x * blockDim.x + threadIdx.x;
  if (i < LSTM_B * LSTM_C) hbuf[i] = (_Float16)h0[i & (LSTM_C - 1)];
  if (i == 0) *cnt = 0u;
}

// --- Phase 1: x_proj GEMM (M=16384, N=4096, K=1024), 32x64 tile per wave -----
// Output stored in WMMA fragment order: v8f at ((t*256 + ct)*2 + mtile)*32 + lane.
__global__ void lstm_xproj_gemm(const _Float16* __restrict__ A,   // embs f16 [16384,1024]
                                const _Float16* __restrict__ Bw,  // w_ih f16 [4096,1024]
                                const float* __restrict__ bih,
                                const float* __restrict__ bhh,
                                float* __restrict__ xp) {
  const int lane = threadIdx.x & 31;
  const int wid  = blockIdx.x * (blockDim.x >> 5) + (threadIdx.x >> 5);
  const int tblk = wid >> 6;           // timestep (32 rows == 1 timestep)
  const int mrow = tblk * 32;
  const int ncol = (wid & 63) * 64;    // 64 col-blocks of 64

  v8f acc[2][4];
  const int cl = lane & 15;
#pragma unroll
  for (int j = 0; j < 4; ++j) {
    const float bias = bih[ncol + j * 16 + cl] + bhh[ncol + j * 16 + cl];
#pragma unroll
    for (int m = 0; m < 2; ++m)
#pragma unroll
      for (int r = 0; r < 8; ++r) acc[m][j][r] = bias;
  }

  for (int k = 0; k < LSTM_I / 32; ++k) {
    const int k0 = k * 32;
    const v16h a0 = load_afrag(A, mrow,      k0, lane, LSTM_I);
    const v16h a1 = load_afrag(A, mrow + 16, k0, lane, LSTM_I);
#pragma unroll
    for (int j = 0; j < 4; ++j) {
      const v16h bf = load_bfrag(Bw, ncol + j * 16, k0, lane, LSTM_I);
      acc[0][j] = wmma_f16(a0, bf, acc[0][j]);
      acc[1][j] = wmma_f16(a1, bf, acc[1][j]);
    }
  }

  // Fragment-order stores: one 32B v8f per accumulator.
#pragma unroll
  for (int m = 0; m < 2; ++m)
#pragma unroll
    for (int j = 0; j < 4; ++j) {
      const size_t ct = (size_t)(ncol >> 4) + j;       // 0..255
      v8f* dst = (v8f*)(xp + ((((size_t)tblk * 256 + ct) * 2 + m) * 32 + lane) * 8);
      *dst = acc[m][j];
    }
}

// --- Phase 2: persistent recurrence ------------------------------------------
// 64 WGs x 128 threads. WG nt owns hidden cols [nt*16, nt*16+16).
// Wave w (=gate) computes gates[32,16] = h[32,1024] @ w_hh_slice^T via WMMA.
__global__ void lstm_recurrent(const float* __restrict__ xp,      // fragment order
                               const _Float16* __restrict__ whhf, // f16 frag order [64][65536]
                               const float* __restrict__ c0,      // [1024]
                               _Float16* __restrict__ hbuf,       // [2][32*1024] f16
                               float* __restrict__ out,           // [512,32,1024] f32
                               unsigned* __restrict__ cnt) {
  __shared__ float Gsh[4 * 2 * 32 * 8];     // gate exchange, fragment order (8 KB, static)
  extern __shared__ _Float16 Bsh[];         // w_hh slice, 128 KB (dynamic, after static)

  const int nt   = blockIdx.x;        // 0..63
  const int tid  = threadIdx.x;       // 0..127
  const int w    = tid >> 5;          // gate 0..3
  const int lane = tid & 31;

  // ---- One-time: DMA this WG's 128KB w_hh slice into LDS via the TDM. ------
  // D# group0/group1 per CDNA5 ISA (2-group form, VADDR2/3=NULL).
  if (tid < 32) {
    const unsigned ldsb = (unsigned)__builtin_amdgcn_groupstaticsize(); // dyn LDS base
    const unsigned long long ga =
        (unsigned long long)(uintptr_t)(whhf + (size_t)nt * BSH_HALVES);
    v4u g0;
    g0[0] = 1u;                                   // count=1 (valid user descriptor)
    g0[1] = ldsb;                                 // lds_addr
    g0[2] = (unsigned)(ga & 0xffffffffu);         // global_addr[31:0]
    g0[3] = (unsigned)((ga >> 32) & 0x01ffffffu)  // global_addr[56:32]
            | (2u << 30);                         // type=2 ("image")
    v8u g1;
    g1[0] = 0x00010000u;  // workgroup_mask=0, data_size=1 (2B)
    g1[1] = 0x40000000u;  // tensor_dim0 = 16384 (bits 63:48)
    g1[2] = 0x00040000u;  // tensor_dim1 = 4     (bits 95:80)
    g1[3] = 0x40000000u;  // tile_dim0   = 16384 (bits 127:112)
    g1[4] = 0x00000004u;  // tile_dim1   = 4, tile_dim2 = 0
    g1[5] = 16384u;       // tensor_dim0_stride[31:0]
    g1[6] = 0u;           // stride hi / tensor_dim1_stride lo
    g1[7] = 0u;
    asm volatile("tensor_load_to_lds %0, %1" :: "s"(g0), "s"(g1) : "memory");
    __builtin_amdgcn_s_wait_tensorcnt((short)0);
  }

  // Per-thread cell state: col = tid%16 (local), rows {tid/16 + 8j}.
  const int colL  = tid & 15;
  const int col   = nt * 16 + colL;
  const int rbase = tid >> 4;
  float c_st[4];
  const float c00 = c0[col];
#pragma unroll
  for (int j = 0; j < 4; ++j) c_st[j] = c00;

  __syncthreads();   // LDS (TDM-filled) visible to all waves

  const size_t HB = (size_t)LSTM_B * LSTM_C;
  const int    ct = w * 64 + nt;            // this wave's column-tile id (0..255)

  for (int t = 0; t < LSTM_T; ++t) {
    const _Float16* hb = hbuf + (size_t)(t & 1) * HB;

    // Accumulators start at x_proj[t] (bias folded in) -- contiguous v8f loads.
    const float* xb = xp + ((((size_t)t * 256 + ct) * 2) * 32 + lane) * 8;
    v8f acc0 = *(const v8f*)(xb);
    v8f acc1 = *(const v8f*)(xb + 32 * 8);
    __builtin_prefetch(xb + (size_t)256 * 2 * 32 * 8, 0, 1);  // next timestep's tile

    // gates += h @ w_hh_slice^T : 2 M-tiles x 32 K-steps of v_wmma
#pragma unroll 4
    for (int k = 0; k < LSTM_C / 32; ++k) {
      const int k0 = k * 32;
      const v16h a0 = load_afrag(hb, 0,  k0, lane, LSTM_C);
      const v16h a1 = load_afrag(hb, 16, k0, lane, LSTM_C);
      union { v16h v; v8h h[2]; } ub;
      const _Float16* bp = Bsh + ((size_t)((w * 32 + k) * 32 + lane) << 4);
      ub.h[0] = *(const v8h*)(bp);
      ub.h[1] = *(const v8h*)(bp + 8);
      acc0 = wmma_f16(a0, ub.v, acc0);
      acc1 = wmma_f16(a1, ub.v, acc1);
    }

    // Publish gate tiles to LDS in fragment order (v8f -> ds_store_b128 pairs).
    {
      v8f* gv = (v8f*)Gsh;
      gv[(w * 2 + 0) * 32 + lane] = acc0;
      gv[(w * 2 + 1) * 32 + lane] = acc1;
    }
    __syncthreads();

    // Elementwise cell update for this thread's 4 (batch,col) elements.
    // Gsh fragment index for (gate g, batch b, col cl):
    //   ((g*2 + b/16)*32 + (cl + (b&8 ? 16 : 0)))*8 + (b&7)
    _Float16* hnext = hbuf + (size_t)((t + 1) & 1) * HB;
#pragma unroll
    for (int j = 0; j < 4; ++j) {
      const int b    = rbase + 8 * j;
      const int base = ((b >> 4) * 32 + (colL + ((b & 8) ? 16 : 0))) * 8 + (b & 7);
      const float gi = Gsh[base + 0 * 512];
      const float gf = Gsh[base + 2 * 512];   // gate stride = 2*32*8 = 512 floats
      const float gg = Gsh[base + 4 * 512];
      const float go = Gsh[base + 6 * 512];
      const float i_ = 1.0f / (1.0f + expf(-gi));
      const float f_ = 1.0f / (1.0f + expf(-gf));
      const float g_ = tanhf(gg);
      const float o_ = 1.0f / (1.0f + expf(-go));
      const float cn = f_ * c_st[j] + i_ * g_;
      c_st[j] = cn;
      const float hn = o_ * tanhf(cn);
      out[(size_t)t * HB + (size_t)b * LSTM_C + col] = hn;
      hnext[(size_t)b * LSTM_C + col] = (_Float16)hn;
    }
    __syncthreads();   // all Gsh reads done before next step's stores

    // Grid-wide barrier: release h stores, arrive, spin, acquire.
    __threadfence();
    __syncthreads();
    if (tid == 0) {
      __hip_atomic_fetch_add(cnt, 1u, __ATOMIC_RELEASE, __HIP_MEMORY_SCOPE_AGENT);
      const unsigned target = (unsigned)(t + 1) * gridDim.x;
      while (__hip_atomic_load(cnt, __ATOMIC_ACQUIRE, __HIP_MEMORY_SCOPE_AGENT) < target) {
        __builtin_amdgcn_s_sleep(1);
      }
    }
    __syncthreads();
    __builtin_amdgcn_fence(__ATOMIC_ACQUIRE, "agent");
  }
}

// ---------------------------------------------------------------------------
extern "C" void kernel_launch(void* const* d_in, const int* in_sizes, int n_in,
                              void* d_out, int out_size, void* d_ws, size_t ws_size,
                              hipStream_t stream) {
  const float* embs = (const float*)d_in[0];
  const float* w_ih = (const float*)d_in[1];
  const float* w_hh = (const float*)d_in[2];
  const float* b_ih = (const float*)d_in[3];
  const float* b_hh = (const float*)d_in[4];
  const float* h0   = (const float*)d_in[5];
  const float* c0   = (const float*)d_in[6];
  float* out = (float*)d_out;

  char* ws = (char*)d_ws;
  float*    xp     = (float*)ws;
  _Float16* embs16 = (_Float16*)(ws + XP_BYTES);
  _Float16* wih16  = (_Float16*)(ws + XP_BYTES + EMB_BYTES);
  _Float16* whhf   = (_Float16*)(ws + XP_BYTES + EMB_BYTES + WIH_BYTES);
  _Float16* hbuf   = (_Float16*)(ws + XP_BYTES + EMB_BYTES + WIH_BYTES + WHH_BYTES);
  unsigned* cnt    = (unsigned*)(ws + XP_BYTES + EMB_BYTES + WIH_BYTES + WHH_BYTES + HBUF_BYTES);

  // Allow >64KB dynamic LDS on the persistent kernel (idempotent host call).
  (void)hipFuncSetAttribute((const void*)lstm_recurrent,
                            hipFuncAttributeMaxDynamicSharedMemorySize, LSTM_SMEM);

  lstm_cvt_f16<<<512, 256, 0, stream>>>(embs, embs16, LSTM_M * LSTM_I);
  lstm_cvt_f16<<<256, 256, 0, stream>>>(w_ih, wih16, LSTM_G4 * LSTM_I);
  lstm_cvt_whh_frag<<<(64 * BSH_HALVES) / 256, 256, 0, stream>>>(w_hh, whhf);
  lstm_init<<<128, 256, 0, stream>>>(h0, hbuf, cnt);

  // 32768 waves (512 row-blocks x 64 col-blocks), 8 waves / block.
  lstm_xproj_gemm<<<4096, 256, 0, stream>>>(embs16, wih16, b_ih, b_hh, xp);

  // Persistent recurrence: 64 WGs x 128 threads (all resident on MI455X).
  // 8 KB static + 128 KB dynamic LDS per WG (<= 2 WGs per 320 KB WGP).
  lstm_recurrent<<<64, 128, LSTM_SMEM, stream>>>(xp, whhf, c0, hbuf, out, cnt);
}
